// GraphNetwork_13219909337179
// MI455X (gfx1250) — matrix operand order repaired and checked by reading, not compiled, not run on hardware
//
#include <hip/hip_runtime.h>
#include <hip/hip_bf16.h>

typedef __attribute__((ext_vector_type(16))) _Float16 v16h;
typedef __attribute__((ext_vector_type(8)))  _Float16 v8h;
typedef __attribute__((ext_vector_type(8)))  float    v8f;

// ---------------------------------------------------------------------------
// Weight convert: W f32 [Kin x Nout] (row-major) -> Wt f16 [NoutPad x Kpad]
// (transposed, zero padded) so each WMMA B-fragment lane reads a contiguous
// K-run of its output column.
// ---------------------------------------------------------------------------
__global__ void __launch_bounds__(256)
k_conv_wt(const float* __restrict__ W, _Float16* __restrict__ Wt,
          int Kin, int Nout, int Kpad, int NoutPad) {
  int idx = blockIdx.x * 256 + threadIdx.x;
  int total = NoutPad * Kpad;
  if (idx >= total) return;
  int n = idx / Kpad;
  int k = idx - n * Kpad;
  float v = (n < Nout && k < Kin) ? W[(size_t)k * Nout + n] : 0.0f;
  Wt[idx] = (_Float16)v;
}

// f32 [M x Kin] -> f16 [M x Kpad], zero padded K
__global__ void __launch_bounds__(256)
k_pack_pad(const float* __restrict__ in, _Float16* __restrict__ out,
           int M, int Kin, int Kpad) {
  int idx = blockIdx.x * 256 + threadIdx.x;
  if (idx >= M * Kpad) return;
  int m = idx / Kpad;
  int k = idx - m * Kpad;
  out[idx] = (k < Kin) ? (_Float16)in[(size_t)m * Kin + k] : (_Float16)0.0f;
}

// ---------------------------------------------------------------------------
// WMMA GEMM, LDS-staged weights, 2-way M register blocking:
//   out[M x Nout] = relu?(A[M x Kpad] * Wt^T + bias)
// Block = 128 threads = 4 waves. The transposed weight matrix
// (NT*16 x Kpad f16, <= 96KB) is staged in LDS once per block. Each wave
// grid-strides over 32-row M-blocks (two 16-row tiles): every B fragment
// pulled from LDS feeds TWO WMMAs (halves LDS traffic per FLOP). B fragments
// for the whole K-chunk are loaded into a register array first so the
// scheduler can pipeline ds_loads ahead of the WMMA stream.
// A fragment: lane m = lane&15, h = lane>>4; halves 0-7 = K h*8..h*8+7,
//             halves 8-15 = K 16+h*8.. (ISA 16-bit A 16x32 layout).
// B fragment: lane n = lane&15; halves 0-7 = K h*8..h*8+7, halves 8-15 =
//             K 16+h*8.. from the transposed-weight row n (same as A).
// D: element r -> row base + h*8 + r, col = t*16 + (lane&15).
// ---------------------------------------------------------------------------
template <int NT>
__global__ void __launch_bounds__(128)
k_gemm(const _Float16* __restrict__ A, const _Float16* __restrict__ Wt,
       const float* __restrict__ bias,
       float* __restrict__ outF, _Float16* __restrict__ outH,
       int M, int Kpad, int Nout, int ldOut, int relu, int mpairs) {
  extern __shared__ _Float16 sW[];          // NT*16 * Kpad halves

  // cooperative weight stage: global -> LDS
  {
    const size_t total = (size_t)NT * 16 * Kpad;        // halves
    for (size_t i = (size_t)threadIdx.x * 8; i < total; i += 128 * 8)
      *(v8h*)(sW + i) = *(const v8h*)(Wt + i);
  }
  __syncthreads();

  const int wave = threadIdx.x >> 5;
  const int lane = threadIdx.x & 31;
  const int l16  = lane & 15;
  const int h    = lane >> 4;

  for (int tp = blockIdx.x * 4 + wave; tp < mpairs; tp += gridDim.x * 4) {
    const _Float16* arow0 = A + (size_t)(tp * 32 + l16) * Kpad;
    const _Float16* arow1 = arow0 + (size_t)16 * Kpad;

    v8f acc0[NT], acc1[NT];
#pragma unroll
    for (int t = 0; t < NT; ++t) { acc0[t] = (v8f){}; acc1[t] = (v8f){}; }

    for (int k = 0; k < Kpad; k += 32) {
      v8h a0lo = *(const v8h*)(arow0 + k + h * 8);
      v8h a0hi = *(const v8h*)(arow0 + k + 16 + h * 8);
      v8h a1lo = *(const v8h*)(arow1 + k + h * 8);
      v8h a1hi = *(const v8h*)(arow1 + k + 16 + h * 8);
      v16h a0 = __builtin_shufflevector(a0lo, a0hi,
                                        0,1,2,3,4,5,6,7,8,9,10,11,12,13,14,15);
      v16h a1 = __builtin_shufflevector(a1lo, a1hi,
                                        0,1,2,3,4,5,6,7,8,9,10,11,12,13,14,15);
      v16h b[NT];
#pragma unroll
      for (int t = 0; t < NT; ++t)
        b[t] = __builtin_shufflevector(
            *(const v8h*)(sW + (size_t)(t * 16 + l16) * Kpad + k + h * 8),
            *(const v8h*)(sW + (size_t)(t * 16 + l16) * Kpad + k + 16 + h * 8),
            0,1,2,3,4,5,6,7,8,9,10,11,12,13,14,15);
#pragma unroll
      for (int t = 0; t < NT; ++t) {
        acc0[t] = __builtin_amdgcn_wmma_f32_16x16x32_f16(
                      false, a0, false, b[t], (short)0, acc0[t], false, false);
        acc1[t] = __builtin_amdgcn_wmma_f32_16x16x32_f16(
                      false, a1, false, b[t], (short)0, acc1[t], false, false);
      }
    }

    const int rbase = tp * 32 + h * 8;
#pragma unroll
    for (int t = 0; t < NT; ++t) {
      int ncol = t * 16 + l16;
      if (ncol < Nout) {
        float bv = bias[ncol];
#pragma unroll
        for (int r = 0; r < 8; ++r) {
          int row0 = rbase + r;
          int row1 = rbase + 16 + r;
          if (row0 < M) {
            float v = acc0[t][r] + bv;
            if (relu) v = fmaxf(v, 0.0f);
            size_t o = (size_t)row0 * ldOut + ncol;
            if (outF) outF[o] = v;
            if (outH) outH[o] = (_Float16)v;
          }
          if (row1 < M) {
            float v = acc1[t][r] + bv;
            if (relu) v = fmaxf(v, 0.0f);
            size_t o = (size_t)row1 * ldOut + ncol;
            if (outF) outF[o] = v;
            if (outH) outH[o] = (_Float16)v;
          }
        }
      }
    }
  }
}

// ---------------------------------------------------------------------------
// LayerNorm over last dim (128). One wave32 per row, 4 elems/lane.
// In-place on f32 x; optional f16 mirror.
// ---------------------------------------------------------------------------
__global__ void __launch_bounds__(256)
k_ln128(float* __restrict__ x, const float* __restrict__ g,
        const float* __restrict__ beta, _Float16* __restrict__ out16, int M) {
  int wave = threadIdx.x >> 5;
  int lane = threadIdx.x & 31;
  int row  = blockIdx.x * 8 + wave;
  if (row >= M) return;
  float* xr = x + (size_t)row * 128;
  float v0[4];
  float s = 0.0f;
#pragma unroll
  for (int i = 0; i < 4; ++i) { v0[i] = xr[lane * 4 + i]; s += v0[i]; }
#pragma unroll
  for (int off = 16; off > 0; off >>= 1) s += __shfl_xor(s, off, 32);
  float mu = s * (1.0f / 128.0f);
  float q = 0.0f;
#pragma unroll
  for (int i = 0; i < 4; ++i) { float d = v0[i] - mu; q += d * d; }
#pragma unroll
  for (int off = 16; off > 0; off >>= 1) q += __shfl_xor(q, off, 32);
  float rstd = rsqrtf(q * (1.0f / 128.0f) + 1e-5f);
#pragma unroll
  for (int i = 0; i < 4; ++i) {
    int c = lane * 4 + i;
    float y = (v0[i] - mu) * rstd * g[c] + beta[c];
    xr[c] = y;
    if (out16) out16[(size_t)row * 128 + c] = (_Float16)y;
  }
}

// Build edge-MLP input: [v16[s[e]], v16[r[e]], ef16[e]]  -> out f16 [E x 384]
// 48 chunks of 8 halves (16B) per edge.
__global__ void __launch_bounds__(256)
k_concat_edge(const _Float16* __restrict__ v16, const int* __restrict__ s,
              const int* __restrict__ r, const _Float16* __restrict__ ef,
              _Float16* __restrict__ out, int E) {
  int idx = blockIdx.x * 256 + threadIdx.x;
  if (idx >= E * 48) return;
  int e = idx / 48, ch = idx - e * 48;
  v8h val;
  if (ch < 16)       val = *(const v8h*)(v16 + (size_t)s[e] * 128 + ch * 8);
  else if (ch < 32)  val = *(const v8h*)(v16 + (size_t)r[e] * 128 + (ch - 16) * 8);
  else               val = *(const v8h*)(ef + (size_t)e * 128 + (ch - 32) * 8);
  *(v8h*)(out + (size_t)e * 384 + ch * 8) = val;
}

// Build node-MLP input: [v16[n], f16(aggm[n]), f16(aggw[n])] -> f16 [N x 384]
__global__ void __launch_bounds__(256)
k_concat_node(const _Float16* __restrict__ v16, const float* __restrict__ am,
              const float* __restrict__ aw, _Float16* __restrict__ out, int Nn) {
  int idx = blockIdx.x * 256 + threadIdx.x;
  if (idx >= Nn * 48) return;
  int n = idx / 48, ch = idx - n * 48;
  _Float16* dst = out + (size_t)n * 384 + ch * 8;
  if (ch < 16) {
    *(v8h*)dst = *(const v8h*)(v16 + (size_t)n * 128 + ch * 8);
  } else {
    const float* src = (ch < 32) ? (am + (size_t)n * 128 + (ch - 16) * 8)
                                 : (aw + (size_t)n * 128 + (ch - 32) * 8);
    v8h t;
#pragma unroll
    for (int i = 0; i < 8; ++i) t[i] = (_Float16)src[i];
    *(v8h*)dst = t;
  }
}

__global__ void __launch_bounds__(256)
k_fill0(float* __restrict__ p, size_t n) {
  size_t i = (size_t)blockIdx.x * 256 + threadIdx.x;
  if (i < n) p[i] = 0.0f;
}

// segment_sum: agg[recv[e]][c] += x[e][c]   (global_atomic_add_f32)
__global__ void __launch_bounds__(256)
k_scatter(const float* __restrict__ x, const int* __restrict__ recv,
          float* __restrict__ agg, int E) {
  int idx = blockIdx.x * 256 + threadIdx.x;
  if (idx >= E * 128) return;
  int e = idx >> 7, c = idx & 127;
  atomicAdd(&agg[(size_t)recv[e] * 128 + c], x[idx]);
}

// residual: acc += d, mirror to f16
__global__ void __launch_bounds__(256)
k_resid(float* __restrict__ acc, const float* __restrict__ d,
        _Float16* __restrict__ o16, size_t n) {
  size_t i = (size_t)blockIdx.x * 256 + threadIdx.x;
  if (i >= n) return;
  float v = acc[i] + d[i];
  acc[i] = v;
  o16[i] = (_Float16)v;
}

// ---------------------------------------------------------------------------
extern "C" void kernel_launch(void* const* d_in, const int* in_sizes, int n_in,
                              void* d_out, int out_size, void* d_ws, size_t ws_size,
                              hipStream_t stream) {
  (void)n_in; (void)out_size; (void)ws_size;
  const int N  = in_sizes[0] / 12;
  const int EM = in_sizes[1] / 7;
  const int EW = in_sizes[2] / 4;

  const float* node_feat  = (const float*)d_in[0];
  const float* mesh_feat  = (const float*)d_in[1];
  const float* world_feat = (const float*)d_in[2];
  const int* mesh_s  = (const int*)d_in[3];
  const int* mesh_r  = (const int*)d_in[4];
  const int* world_s = (const int*)d_in[5];
  const int* world_r = (const int*)d_in[6];

  // ---- params: jax pytree flatten order (dict keys sorted) ----
  int pi = 7;
  auto nxt = [&]() { return (const float*)d_in[pi++]; };
  struct MLPP { const float* b[4]; const float* beta; const float* g; const float* w[4]; };
  auto readLN = [&]() {
    MLPP m;
    for (int i = 0; i < 4; ++i) m.b[i] = nxt();
    m.beta = nxt(); m.g = nxt();
    for (int i = 0; i < 4; ++i) m.w[i] = nxt();
    return m;
  };
  MLPP dec{};
  for (int i = 0; i < 4; ++i) dec.b[i] = nxt();
  for (int i = 0; i < 4; ++i) dec.w[i] = nxt();
  dec.g = nullptr; dec.beta = nullptr;
  MLPP enc_mesh  = readLN();
  MLPP enc_node  = readLN();
  MLPP enc_world = readLN();
  MLPP st_mesh[18], st_node[18], st_world[18];
  for (int s = 0; s < 18; ++s) {        // steps[i] keys: mesh < node < world
    st_mesh[s] = readLN(); st_node[s] = readLN(); st_world[s] = readLN();
  }

  // ---- workspace carve (deterministic) ----
  char* cur = (char*)d_ws;
  auto carve = [&](size_t bytes) -> void* {
    void* ret = (void*)cur;
    cur += (bytes + 255) & ~(size_t)255;
    return ret;
  };
  float*    v32   = (float*)carve((size_t)N  * 128 * 4);
  _Float16* v16b  = (_Float16*)carve((size_t)N  * 128 * 2);
  float*    em32  = (float*)carve((size_t)EM * 128 * 4);
  _Float16* em16  = (_Float16*)carve((size_t)EM * 128 * 2);
  float*    ew32  = (float*)carve((size_t)EW * 128 * 4);
  _Float16* ew16  = (_Float16*)carve((size_t)EW * 128 * 2);
  float*    aggm  = (float*)carve((size_t)N  * 128 * 4);
  float*    aggw  = (float*)carve((size_t)N  * 128 * 4);
  float*    nem   = (float*)carve((size_t)EM * 128 * 4);
  float*    n_ew  = (float*)carve((size_t)EW * 128 * 4);
  float*    n_v   = (float*)carve((size_t)N  * 128 * 4);
  _Float16* h1    = (_Float16*)carve((size_t)EM * 128 * 2);
  _Float16* h2    = (_Float16*)carve((size_t)EM * 128 * 2);
  _Float16* cat   = (_Float16*)carve((size_t)EM * 384 * 2);
  _Float16* packb = (_Float16*)carve((size_t)EM * 32 * 2);

  struct LayerD { const float* W; const float* b; int Kin, Nout, Kpad, NoutPad; _Float16* Wt; };
  auto mkL = [&](const float* W, const float* b, int Kin, int Nout) {
    LayerD L; L.W = W; L.b = b; L.Kin = Kin; L.Nout = Nout;
    L.Kpad = (Kin + 31) & ~31; L.NoutPad = (Nout + 15) & ~15;
    L.Wt = (_Float16*)carve((size_t)L.NoutPad * L.Kpad * 2);
    return L;
  };
  struct MLPD { LayerD L[4]; const float* g; const float* beta; };
  auto mkMLP = [&](const MLPP& m, int Kin0, int NoutLast) {
    MLPD d;
    d.L[0] = mkL(m.w[0], m.b[0], Kin0, 128);
    d.L[1] = mkL(m.w[1], m.b[1], 128, 128);
    d.L[2] = mkL(m.w[2], m.b[2], 128, 128);
    d.L[3] = mkL(m.w[3], m.b[3], 128, NoutLast);
    d.g = m.g; d.beta = m.beta;
    return d;
  };
  MLPD dEncNode  = mkMLP(enc_node, 12, 128);
  MLPD dEncMesh  = mkMLP(enc_mesh, 7, 128);
  MLPD dEncWorld = mkMLP(enc_world, 4, 128);
  MLPD dDec      = mkMLP(dec, 128, 3);
  MLPD dStepM[18], dStepW[18], dStepN[18];
  for (int s = 0; s < 18; ++s) {
    dStepM[s] = mkMLP(st_mesh[s], 384, 128);
    dStepW[s] = mkMLP(st_world[s], 384, 128);
    dStepN[s] = mkMLP(st_node[s], 384, 128);
  }

  // ---- convert all weights to transposed f16 ----
  auto convMLP = [&](MLPD& d) {
    for (int i = 0; i < 4; ++i) {
      LayerD& L = d.L[i];
      int total = L.NoutPad * L.Kpad;
      k_conv_wt<<<(total + 255) / 256, 256, 0, stream>>>(
          L.W, L.Wt, L.Kin, L.Nout, L.Kpad, L.NoutPad);
    }
  };
  convMLP(dEncNode); convMLP(dEncMesh); convMLP(dEncWorld); convMLP(dDec);
  for (int s = 0; s < 18; ++s) { convMLP(dStepM[s]); convMLP(dStepW[s]); convMLP(dStepN[s]); }

  auto gemm = [&](const _Float16* A, LayerD& L, int M, float* oF, _Float16* oH,
                  int ldOut, int relu) {
    int mtiles = (M + 15) / 16;
    int mpairs = (mtiles + 1) / 2;           // 32-row blocks per wave
    int blocks = (mpairs + 3) / 4;
    if (blocks > 640) blocks = 640;          // grid-stride: amortize LDS stage
    size_t lds = (size_t)L.NoutPad * L.Kpad * 2;
    if (L.NoutPad == 128)
      k_gemm<8><<<dim3(blocks), 128, lds, stream>>>(
          A, L.Wt, L.b, oF, oH, M, L.Kpad, L.Nout, ldOut, relu, mpairs);
    else
      k_gemm<1><<<dim3(blocks), 128, lds, stream>>>(
          A, L.Wt, L.b, oF, oH, M, L.Kpad, L.Nout, ldOut, relu, mpairs);
  };
  auto runMLP = [&](const _Float16* A0, int M, MLPD& d, float* outF, _Float16* outH16) {
    gemm(A0, d.L[0], M, nullptr, h1, 128, 1);
    gemm(h1, d.L[1], M, nullptr, h2, 128, 1);
    gemm(h2, d.L[2], M, nullptr, h1, 128, 1);
    gemm(h1, d.L[3], M, outF, nullptr, 128, 0);
    if (d.g)
      k_ln128<<<(M + 7) / 8, 256, 0, stream>>>(outF, d.g, d.beta, outH16, M);
  };

  // ---- encoders ----
  k_pack_pad<<<(N * 32 + 255) / 256, 256, 0, stream>>>(node_feat, packb, N, 12, 32);
  runMLP(packb, N, dEncNode, v32, v16b);
  k_pack_pad<<<(EM * 32 + 255) / 256, 256, 0, stream>>>(mesh_feat, packb, EM, 7, 32);
  runMLP(packb, EM, dEncMesh, em32, em16);
  k_pack_pad<<<(EW * 32 + 255) / 256, 256, 0, stream>>>(world_feat, packb, EW, 4, 32);
  runMLP(packb, EW, dEncWorld, ew32, ew16);

  // ---- 18 processor steps ----
  for (int s = 0; s < 18; ++s) {
    k_concat_edge<<<(EM * 48 + 255) / 256, 256, 0, stream>>>(v16b, mesh_s, mesh_r, em16, cat, EM);
    runMLP(cat, EM, dStepM[s], nem, nullptr);
    k_concat_edge<<<(EW * 48 + 255) / 256, 256, 0, stream>>>(v16b, world_s, world_r, ew16, cat, EW);
    runMLP(cat, EW, dStepW[s], n_ew, nullptr);
    k_fill0<<<((size_t)N * 128 + 255) / 256, 256, 0, stream>>>(aggm, (size_t)N * 128);
    k_fill0<<<((size_t)N * 128 + 255) / 256, 256, 0, stream>>>(aggw, (size_t)N * 128);
    k_scatter<<<(EM * 128 + 255) / 256, 256, 0, stream>>>(nem, mesh_r, aggm, EM);
    k_scatter<<<(EW * 128 + 255) / 256, 256, 0, stream>>>(n_ew, world_r, aggw, EW);
    k_concat_node<<<(N * 48 + 255) / 256, 256, 0, stream>>>(v16b, aggm, aggw, cat, N);
    runMLP(cat, N, dStepN[s], n_v, nullptr);
    k_resid<<<((size_t)N * 128 + 255) / 256, 256, 0, stream>>>(v32, n_v, v16b, (size_t)N * 128);
    k_resid<<<((size_t)EM * 128 + 255) / 256, 256, 0, stream>>>(em32, nem, em16, (size_t)EM * 128);
    k_resid<<<((size_t)EW * 128 + 255) / 256, 256, 0, stream>>>(ew32, n_ew, ew16, (size_t)EW * 128);
  }

  // ---- decoder ----
  gemm(v16b, dDec.L[0], N, nullptr, h1, 128, 1);
  gemm(h1,  dDec.L[1], N, nullptr, h2, 128, 1);
  gemm(h2,  dDec.L[2], N, nullptr, h1, 128, 1);
  gemm(h1,  dDec.L[3], N, (float*)d_out, nullptr, 3, 0);
}
